// ModularLinear_59768764891495
// MI455X (gfx1250) — matrix operand order repaired and checked by reading, not compiled, not run on hardware
//
#include <hip/hip_runtime.h>

// ---------------------------------------------------------------------------
// ModularLinear (MoE-style selected expert GEMM) for gfx1250 / MI455X.
//   out[t,b,k*OUT+o] = sum_i x[t,b,i] * W[sel[b,k],i,o] + bias[sel[b,k],o]
// fp32 in/out via split-bf16 (hi+lo) 3-mult WMMA: A*B ~= AhBh + AhBl + AlBh.
// Phase 1 (once): convert x and W into bf16 hi/lo planes in d_ws; W is also
//   pre-transposed to [e][o][i] so GEMM staging is a straight k-contiguous copy.
// Phase 2: double-buffered WMMA GEMM with zero conversion VALU in the loop.
// Falls back to a fused single-kernel version if d_ws is too small.
// ---------------------------------------------------------------------------

typedef __attribute__((ext_vector_type(16))) __bf16        v16bf;
typedef __attribute__((ext_vector_type(8)))  __bf16        v8bf;
typedef __attribute__((ext_vector_type(8)))  float         v8f;
typedef __attribute__((ext_vector_type(4)))  float         f32x4;
typedef __attribute__((ext_vector_type(2)))  unsigned int  u32x2;
typedef __attribute__((ext_vector_type(4)))  unsigned int  u32x4;

constexpr int T_DIM   = 1024;
constexpr int B_DIM   = 8;
constexpr int IN_DIM  = 1024;
constexpr int OUT_DIM = 1024;
constexpr int K_SEL   = 2;
constexpr int N_MOD   = 16;

constexpr int BM = 128;      // time rows per block
constexpr int BN = 128;      // out cols per block
constexpr int BK = 32;       // K step (bf16 WMMA depth)
constexpr int LDS_ROW = 80;  // bytes/row: 64 used + 16 pad (16B-aligned, 20-bank stride)
constexpr int PLANE   = BM * LDS_ROW;   // 10240 B per bf16 plane
constexpr int BUF     = 4 * PLANE;      // Ah|Al|Bh|Bl = 40 KB per buffer

constexpr size_t X_ELEMS = (size_t)T_DIM * B_DIM * IN_DIM;    // 8M
constexpr size_t W_ELEMS = (size_t)N_MOD * IN_DIM * OUT_DIM;  // 16M

struct PackedPair { unsigned hi, lo; };

// Split two floats into packed bf16 hi pair and residual-lo pair (RNE both).
__device__ __forceinline__ PackedPair split_pack2(float f0, float f1) {
  const unsigned u0 = __float_as_uint(f0);
  const unsigned u1 = __float_as_uint(f1);
  const unsigned r0 = u0 + 0x7FFFu + ((u0 >> 16) & 1u);
  const unsigned r1 = u1 + 0x7FFFu + ((u1 >> 16) & 1u);
  const float g0 = f0 - __uint_as_float(r0 & 0xFFFF0000u);
  const float g1 = f1 - __uint_as_float(r1 & 0xFFFF0000u);
  const unsigned s0 = __float_as_uint(g0);
  const unsigned s1 = __float_as_uint(g1);
  const unsigned t0 = s0 + 0x7FFFu + ((s0 >> 16) & 1u);
  const unsigned t1 = s1 + 0x7FFFu + ((s1 >> 16) & 1u);
  PackedPair out;
  out.hi = __builtin_amdgcn_perm(r1, r0, 0x07060302u);   // {bf16(f1),bf16(f0)}
  out.lo = __builtin_amdgcn_perm(t1, t0, 0x07060302u);
  return out;
}

__device__ __forceinline__ v16bf cat8(v8bf a, v8bf b) {
  return __builtin_shufflevector(a, b, 0, 1, 2, 3, 4, 5, 6, 7,
                                       8, 9, 10, 11, 12, 13, 14, 15);
}

// ---------------------------------------------------------------------------
// Phase-1a: x [t][b][i] f32 -> Xh/Xl bf16 planes, same layout.
// ---------------------------------------------------------------------------
__global__ __launch_bounds__(256)
void convert_x_kernel(const float* __restrict__ x,
                      unsigned short* __restrict__ xh,
                      unsigned short* __restrict__ xl) {
  const size_t i4 = ((size_t)blockIdx.x * 256 + threadIdx.x) * 4;
  const f32x4 v = *(const f32x4*)(x + i4);
  const PackedPair p01 = split_pack2(v[0], v[1]);
  const PackedPair p23 = split_pack2(v[2], v[3]);
  u32x2 hv, lv;
  hv.x = p01.hi; hv.y = p23.hi;
  lv.x = p01.lo; lv.y = p23.lo;
  *(u32x2*)(xh + i4) = hv;
  *(u32x2*)(xl + i4) = lv;
}

// ---------------------------------------------------------------------------
// Phase-1b: W [e][i][o] f32 -> Wth/Wtl bf16 planes transposed to [e][o][i].
// 64x64 tile per block through padded LDS.
// ---------------------------------------------------------------------------
__global__ __launch_bounds__(256)
void convert_w_kernel(const float* __restrict__ w,
                      unsigned short* __restrict__ wth,
                      unsigned short* __restrict__ wtl) {
  __shared__ unsigned short lh[64 * 72];   // [o][i], row stride 72 (144B, 16B-aligned)
  __shared__ unsigned short ll[64 * 72];

  const int o0 = blockIdx.x * 64;
  const int i0 = blockIdx.y * 64;
  const int e  = blockIdx.z;
  const int tid = threadIdx.x;

  const int r0 = tid >> 4;            // 0..15
  const int c4 = (tid & 15) * 4;      // o within tile: 0,4,..,60
  const float* src = w + ((size_t)e * IN_DIM + i0) * OUT_DIM + o0;

#pragma unroll
  for (int p = 0; p < 4; ++p) {
    const int r = p * 16 + r0;        // i within tile
    const f32x4 v = *(const f32x4*)(src + (size_t)r * OUT_DIM + c4);
#pragma unroll
    for (int j = 0; j < 2; ++j) {
      const PackedPair pp = split_pack2(v[2 * j], v[2 * j + 1]);
      lh[(c4 + 2 * j)     * 72 + r] = (unsigned short)(pp.hi & 0xFFFFu);
      lh[(c4 + 2 * j + 1) * 72 + r] = (unsigned short)(pp.hi >> 16);
      ll[(c4 + 2 * j)     * 72 + r] = (unsigned short)(pp.lo & 0xFFFFu);
      ll[(c4 + 2 * j + 1) * 72 + r] = (unsigned short)(pp.lo >> 16);
    }
  }
  __syncthreads();

  const int ro  = tid >> 2;           // o row 0..63
  const int seg = (tid & 3) * 16;     // 16 ushorts = 32B segment of the i-row
  unsigned short* dh = wth + ((size_t)(e * OUT_DIM + o0 + ro)) * IN_DIM + i0 + seg;
  unsigned short* dl = wtl + ((size_t)(e * OUT_DIM + o0 + ro)) * IN_DIM + i0 + seg;
  *(u32x4*)dh       = *(const u32x4*)(lh + ro * 72 + seg);
  *(u32x4*)(dh + 8) = *(const u32x4*)(lh + ro * 72 + seg + 8);
  *(u32x4*)dl       = *(const u32x4*)(ll + ro * 72 + seg);
  *(u32x4*)(dl + 8) = *(const u32x4*)(ll + ro * 72 + seg + 8);
}

// ---------------------------------------------------------------------------
// Phase-2: WMMA GEMM from pre-split bf16 planes. Staging = pure b128 copies.
// ---------------------------------------------------------------------------
__global__ __launch_bounds__(256)
void gemm_bf16_planes(const unsigned short* __restrict__ xh,
                      const unsigned short* __restrict__ xl,
                      const unsigned short* __restrict__ wth,
                      const unsigned short* __restrict__ wtl,
                      const int*   __restrict__ sel,
                      const float* __restrict__ bias,
                      float*       __restrict__ out) {
  __shared__ unsigned char smem[2 * BUF];   // 80 KB double buffer

  const int tn = blockIdx.x;
  const int tm = blockIdx.y;
  const int bk = blockIdx.z;
  const int b  = bk / K_SEL;
  const int ks = bk % K_SEL;
  const int e  = sel[bk] & (N_MOD - 1);

  const int tid  = threadIdx.x;
  const int lane = tid & 31;
  const int wave = tid >> 5;
  const int wn   = wave & 3;
  const int wm   = wave >> 2;
  const int half = lane >> 4;
  const int l16  = lane & 15;

  // staging: 2 threads per 64B row; cs in ushort units (16 ushorts = 32B)
  const int cr = tid >> 1;            // 0..127
  const int cs = (tid & 1) * 16;

  const unsigned short* aH = xh + ((size_t)((tm * BM + cr) * B_DIM + b)) * IN_DIM + cs;
  const unsigned short* aL = xl + ((size_t)((tm * BM + cr) * B_DIM + b)) * IN_DIM + cs;
  const unsigned short* bH = wth + ((size_t)(e * OUT_DIM + tn * BN + cr)) * IN_DIM + cs;
  const unsigned short* bL = wtl + ((size_t)(e * OUT_DIM + tn * BN + cr)) * IN_DIM + cs;

  u32x4 rA[2][2], rB[2][2];           // [plane][16B chunk]
  auto load_tiles = [&](int k0) {
    rA[0][0] = *(const u32x4*)(aH + k0);
    rA[0][1] = *(const u32x4*)(aH + k0 + 8);
    rA[1][0] = *(const u32x4*)(aL + k0);
    rA[1][1] = *(const u32x4*)(aL + k0 + 8);
    rB[0][0] = *(const u32x4*)(bH + k0);
    rB[0][1] = *(const u32x4*)(bH + k0 + 8);
    rB[1][0] = *(const u32x4*)(bL + k0);
    rB[1][1] = *(const u32x4*)(bL + k0 + 8);
  };
  auto store_tiles = [&](int buf) {
    unsigned char* base = smem + buf * BUF;
    const int off = cr * LDS_ROW + cs * 2;   // bytes
    *(u32x4*)(base + off)                  = rA[0][0];
    *(u32x4*)(base + off + 16)             = rA[0][1];
    *(u32x4*)(base + PLANE + off)          = rA[1][0];
    *(u32x4*)(base + PLANE + off + 16)     = rA[1][1];
    *(u32x4*)(base + 2 * PLANE + off)      = rB[0][0];
    *(u32x4*)(base + 2 * PLANE + off + 16) = rB[0][1];
    *(u32x4*)(base + 3 * PLANE + off)      = rB[1][0];
    *(u32x4*)(base + 3 * PLANE + off + 16) = rB[1][1];
  };

  v8f acc[4][2];
#pragma unroll
  for (int mi = 0; mi < 4; ++mi)
#pragma unroll
    for (int ni = 0; ni < 2; ++ni)
      acc[mi][ni] = (v8f)(0.0f);

  load_tiles(0);
  store_tiles(0);
  __syncthreads();

  int cur = 0;
  for (int k0 = 0; k0 < IN_DIM; k0 += BK) {
    const bool has_next = (k0 + BK) < IN_DIM;
    if (has_next) load_tiles(k0 + BK);   // global loads in flight during WMMA

    const unsigned char* Ah = smem + cur * BUF;
    const unsigned char* Al = Ah + PLANE;
    const unsigned char* Bh = Ah + 2 * PLANE;
    const unsigned char* Bl = Ah + 3 * PLANE;

    // fragments (layouts per CDNA5 ISA 7.12.2)
    v16bf Af[4][2], Bf[2][2];
#pragma unroll
    for (int mi = 0; mi < 4; ++mi) {
      const int base = (wm * 64 + mi * 16 + l16) * LDS_ROW + half * 16;
      Af[mi][0] = cat8(*(const v8bf*)(Ah + base), *(const v8bf*)(Ah + base + 32));
      Af[mi][1] = cat8(*(const v8bf*)(Al + base), *(const v8bf*)(Al + base + 32));
    }
#pragma unroll
    for (int ni = 0; ni < 2; ++ni) {
      const int base = (wn * 32 + ni * 16 + l16) * LDS_ROW + half * 16;
      Bf[ni][0] = cat8(*(const v8bf*)(Bh + base), *(const v8bf*)(Bh + base + 32));
      Bf[ni][1] = cat8(*(const v8bf*)(Bl + base), *(const v8bf*)(Bl + base + 32));
    }

    // 3-mult split-bf16 WMMA: AhBh + AhBl + AlBh
#pragma unroll
    for (int mi = 0; mi < 4; ++mi)
#pragma unroll
      for (int ni = 0; ni < 2; ++ni) {
        acc[mi][ni] = __builtin_amdgcn_wmma_f32_16x16x32_bf16(
            false, Af[mi][0], false, Bf[ni][0], (short)0, acc[mi][ni], false, false);
        acc[mi][ni] = __builtin_amdgcn_wmma_f32_16x16x32_bf16(
            false, Af[mi][0], false, Bf[ni][1], (short)0, acc[mi][ni], false, false);
        acc[mi][ni] = __builtin_amdgcn_wmma_f32_16x16x32_bf16(
            false, Af[mi][1], false, Bf[ni][0], (short)0, acc[mi][ni], false, false);
      }

    if (has_next) store_tiles(cur ^ 1);
    __syncthreads();
    cur ^= 1;
  }

  // epilogue: bias + scatter per C/D layout (VGPR r -> M = r + 8*half)
  float biasv[2];
#pragma unroll
  for (int ni = 0; ni < 2; ++ni)
    biasv[ni] = bias[e * OUT_DIM + tn * BN + wn * 32 + ni * 16 + l16];

#pragma unroll
  for (int mi = 0; mi < 4; ++mi)
#pragma unroll
    for (int ni = 0; ni < 2; ++ni) {
      const int n = tn * BN + wn * 32 + ni * 16 + l16;
#pragma unroll
      for (int r = 0; r < 8; ++r) {
        const int m = tm * BM + wm * 64 + mi * 16 + half * 8 + r;
        const size_t idx =
            ((size_t)(m * B_DIM + b)) * (K_SEL * OUT_DIM) + (size_t)ks * OUT_DIM + n;
        out[idx] = acc[mi][ni][r] + biasv[ni];
      }
    }
}

// ---------------------------------------------------------------------------
// Fallback: fused single-kernel version (converts in the GEMM loop).
// ---------------------------------------------------------------------------
__global__ __launch_bounds__(256)
void modular_linear_fused(const float* __restrict__ x,
                          const int*   __restrict__ sel,
                          const float* __restrict__ weight,
                          const float* __restrict__ bias,
                          float*       __restrict__ out) {
  __shared__ unsigned char smem[2 * BUF];

  const int tn = blockIdx.x;
  const int tm = blockIdx.y;
  const int bk = blockIdx.z;
  const int b  = bk / K_SEL;
  const int ks = bk % K_SEL;
  const int e  = sel[bk] & (N_MOD - 1);

  const int tid  = threadIdx.x;
  const int lane = tid & 31;
  const int wave = tid >> 5;
  const int wn   = wave & 3;
  const int wm   = wave >> 2;
  const int half = lane >> 4;
  const int l16  = lane & 15;

  const int ar  = tid >> 3;
  const int ac  = (tid & 7) * 4;
  const int kk0 = (tid >> 5) * 4;
  const int bn0 = (tid & 31) * 4;

  const float* wbase = weight + (size_t)e * IN_DIM * OUT_DIM;
  const float* aptr  = x + ((size_t)((tm * BM + ar) * B_DIM + b)) * IN_DIM + ac;
  const float* bptr  = wbase + (size_t)kk0 * OUT_DIM + tn * BN + bn0;

  f32x4 aReg[4], bReg[4];
  auto load_tiles = [&](int k0) {
#pragma unroll
    for (int p = 0; p < 4; ++p)
      aReg[p] = *(const f32x4*)(aptr + (size_t)p * 32 * B_DIM * IN_DIM + k0);
#pragma unroll
    for (int q = 0; q < 4; ++q)
      bReg[q] = *(const f32x4*)(bptr + (size_t)(k0 + q) * OUT_DIM);
  };
  auto store_tiles = [&](int buf) {
    unsigned char* Ah = smem + buf * BUF;
    unsigned char* Al = Ah + PLANE;
    unsigned char* Bh = Ah + 2 * PLANE;
    unsigned char* Bl = Ah + 3 * PLANE;
#pragma unroll
    for (int p = 0; p < 4; ++p) {
      const PackedPair p01 = split_pack2(aReg[p][0], aReg[p][1]);
      const PackedPair p23 = split_pack2(aReg[p][2], aReg[p][3]);
      u32x2 hv, lv;
      hv.x = p01.hi; hv.y = p23.hi;
      lv.x = p01.lo; lv.y = p23.lo;
      *(u32x2*)(Ah + (p * 32 + ar) * LDS_ROW + ac * 2) = hv;
      *(u32x2*)(Al + (p * 32 + ar) * LDS_ROW + ac * 2) = lv;
    }
#pragma unroll
    for (int j = 0; j < 4; ++j) {
      const PackedPair q01 = split_pack2(bReg[0][j], bReg[1][j]);
      const PackedPair q23 = split_pack2(bReg[2][j], bReg[3][j]);
      u32x2 hv, lv;
      hv.x = q01.hi; hv.y = q23.hi;
      lv.x = q01.lo; lv.y = q23.lo;
      *(u32x2*)(Bh + (bn0 + j) * LDS_ROW + kk0 * 2) = hv;
      *(u32x2*)(Bl + (bn0 + j) * LDS_ROW + kk0 * 2) = lv;
    }
  };

  v8f acc[4][2];
#pragma unroll
  for (int mi = 0; mi < 4; ++mi)
#pragma unroll
    for (int ni = 0; ni < 2; ++ni)
      acc[mi][ni] = (v8f)(0.0f);

  load_tiles(0);
  store_tiles(0);
  __syncthreads();

  int cur = 0;
  for (int k0 = 0; k0 < IN_DIM; k0 += BK) {
    const bool has_next = (k0 + BK) < IN_DIM;
    if (has_next) load_tiles(k0 + BK);

    const unsigned char* Ah = smem + cur * BUF;
    const unsigned char* Al = Ah + PLANE;
    const unsigned char* Bh = Ah + 2 * PLANE;
    const unsigned char* Bl = Ah + 3 * PLANE;

    v16bf Af[4][2], Bf[2][2];
#pragma unroll
    for (int mi = 0; mi < 4; ++mi) {
      const int base = (wm * 64 + mi * 16 + l16) * LDS_ROW + half * 16;
      Af[mi][0] = cat8(*(const v8bf*)(Ah + base), *(const v8bf*)(Ah + base + 32));
      Af[mi][1] = cat8(*(const v8bf*)(Al + base), *(const v8bf*)(Al + base + 32));
    }
#pragma unroll
    for (int ni = 0; ni < 2; ++ni) {
      const int base = (wn * 32 + ni * 16 + l16) * LDS_ROW + half * 16;
      Bf[ni][0] = cat8(*(const v8bf*)(Bh + base), *(const v8bf*)(Bh + base + 32));
      Bf[ni][1] = cat8(*(const v8bf*)(Bl + base), *(const v8bf*)(Bl + base + 32));
    }

#pragma unroll
    for (int mi = 0; mi < 4; ++mi)
#pragma unroll
      for (int ni = 0; ni < 2; ++ni) {
        acc[mi][ni] = __builtin_amdgcn_wmma_f32_16x16x32_bf16(
            false, Af[mi][0], false, Bf[ni][0], (short)0, acc[mi][ni], false, false);
        acc[mi][ni] = __builtin_amdgcn_wmma_f32_16x16x32_bf16(
            false, Af[mi][0], false, Bf[ni][1], (short)0, acc[mi][ni], false, false);
        acc[mi][ni] = __builtin_amdgcn_wmma_f32_16x16x32_bf16(
            false, Af[mi][1], false, Bf[ni][0], (short)0, acc[mi][ni], false, false);
      }

    if (has_next) store_tiles(cur ^ 1);
    __syncthreads();
    cur ^= 1;
  }

  float biasv[2];
#pragma unroll
  for (int ni = 0; ni < 2; ++ni)
    biasv[ni] = bias[e * OUT_DIM + tn * BN + wn * 32 + ni * 16 + l16];

#pragma unroll
  for (int mi = 0; mi < 4; ++mi)
#pragma unroll
    for (int ni = 0; ni < 2; ++ni) {
      const int n = tn * BN + wn * 32 + ni * 16 + l16;
#pragma unroll
      for (int r = 0; r < 8; ++r) {
        const int m = tm * BM + wm * 64 + mi * 16 + half * 8 + r;
        const size_t idx =
            ((size_t)(m * B_DIM + b)) * (K_SEL * OUT_DIM) + (size_t)ks * OUT_DIM + n;
        out[idx] = acc[mi][ni][r] + biasv[ni];
      }
    }
}

extern "C" void kernel_launch(void* const* d_in, const int* in_sizes, int n_in,
                              void* d_out, int out_size, void* d_ws, size_t ws_size,
                              hipStream_t stream) {
  (void)in_sizes; (void)n_in; (void)out_size;
  const float* x      = (const float*)d_in[0];
  const int*   sel    = (const int*)d_in[1];
  const float* weight = (const float*)d_in[2];
  const float* bias   = (const float*)d_in[3];
  float*       out    = (float*)d_out;

  const dim3 grid(OUT_DIM / BN, T_DIM / BM, B_DIM * K_SEL);  // (8, 8, 16)
  const size_t need = (X_ELEMS + W_ELEMS) * 2 /*hi+lo*/ * 2 /*bytes*/;  // 96 MB

  if (ws_size >= need) {
    unsigned short* xh  = (unsigned short*)d_ws;
    unsigned short* xl  = xh + X_ELEMS;
    unsigned short* wth = xl + X_ELEMS;
    unsigned short* wtl = wth + W_ELEMS;

    convert_x_kernel<<<(unsigned)(X_ELEMS / 1024), dim3(256), 0, stream>>>(x, xh, xl);
    convert_w_kernel<<<dim3(OUT_DIM / 64, IN_DIM / 64, N_MOD), dim3(256), 0, stream>>>(
        weight, wth, wtl);
    gemm_bf16_planes<<<grid, dim3(256), 0, stream>>>(xh, xl, wth, wtl, sel, bias, out);
  } else {
    modular_linear_fused<<<grid, dim3(256), 0, stream>>>(x, sel, weight, bias, out);
  }
}
